// half_dim_bi_36344013258825
// MI455X (gfx1250) — matrix operand order repaired and checked by reading, not compiled, not run on hardware
//
#include <hip/hip_runtime.h>

typedef __attribute__((ext_vector_type(16))) _Float16 v16h;
typedef __attribute__((ext_vector_type(8)))  float    v8f;
typedef __attribute__((ext_vector_type(4)))  _Float16 v4h;

#define LLEN    8192
#define NBLK    512          // LLEN / 16 time blocks
#define BATCH   4
#define DMODEL  1024
#define HALFD   512
#define THREADS 512
#define ROUNDS  64           // 64 rounds * 4 substeps * 2 lag-blocks = 512 lag blocks

// Triangular-Toeplitz conv as WMMA GEMM chains (see round-0 derivation).
// x panel and A tiles live in LDS with a 16B-chunk swizzle that makes every
// wave32 ds_load_b128 a perfect bank permutation per 16-lane phase.
__global__ __launch_bounds__(THREADS, 1)
void hyena_tconv_wmma(const float* __restrict__ x,
                      const float* __restrict__ kk,
                      const float* __restrict__ Dg,
                      float* __restrict__ out)
{
  // x panel: block (J, b) = 16 halves; block id q = 4*J + b; chunk half c in {0,1}
  // byte offset = ((q&15)<<4) + ((q>>4)<<9) + (c<<8)   -> bank-conflict-free
  __shared__ alignas(256) unsigned char xarena[BATCH * LLEN * 2];  // 64 KB
  __shared__ alignas(32)  _Float16 kh[LLEN];                       // 16 KB
  // A tiles: lane L's chunk c at ((L&15)<<4) + ((L>>4)<<9) + (c<<8) (+ s<<10)
  __shared__ alignas(256) unsigned char aarena[4 * 1024];          //  4 KB
  __shared__ float stage[16 * 256];                                // 16 KB

  const int wg   = blockIdx.x;
  const int d    = wg >> 1;          // channel
  const int pi   = wg & 1;           // which half of the 128 tile-groups
  const int tid  = threadIdx.x;
  const int lane = tid & 31;
  const int wv   = tid >> 5;
  const bool causal = (d >= HALFD);  // second half of channels: causal conv

  const size_t chan_off = (size_t)d * LLEN;
  const size_t bstride  = (size_t)DMODEL * LLEN;

  // ---------------- prologue: stage x (4x8192) and k into LDS as f16 --------
  #pragma unroll
  for (int b = 0; b < BATCH; ++b) {
    const float4* src = (const float4*)(x + (size_t)b * bstride + chan_off);
    #pragma unroll
    for (int r = 0; r < 4; ++r) {
      int idx = tid + THREADS * r;                 // float4 index 0..2047
      float4 v = src[idx];
      v4h hv = { (_Float16)v.x, (_Float16)v.y, (_Float16)v.z, (_Float16)v.w };
      int t0 = idx * 4;                            // time of first half
      int J  = t0 >> 4;
      int q  = (J << 2) | b;
      int a  = ((q & 15) << 4) + ((q >> 4) << 9)   // swizzled chunk base
             + (((t0 >> 3) & 1) << 8)              // chunk half
             + ((t0 & 7) << 1);                    // byte within chunk (0 or 8)
      *(v4h*)(xarena + a) = hv;
    }
  }
  {
    const float4* ks = (const float4*)(kk + chan_off);
    #pragma unroll
    for (int r = 0; r < 4; ++r) {
      int idx = tid + THREADS * r;
      float4 v = ks[idx];
      v4h hv = { (_Float16)v.x, (_Float16)v.y, (_Float16)v.z, (_Float16)v.w };
      *(v4h*)&kh[idx * 4] = hv;
    }
  }

  // wave's strided tile-group list (ascending; balances the triangle)
  int groups[4];
  #pragma unroll
  for (int j = 0; j < 4; ++j) groups[j] = pi + 2 * (wv + 16 * j);

  v8f acc[4];
  #pragma unroll
  for (int j = 0; j < 4; ++j) acc[j] = (v8f)0.0f;

  // A-builder constants: thread -> (lane, element) of the 16x32 f16 A layout.
  // element e of v16h at lane L holds K = e + (L&16) (contiguous order, same as B), M = L&15.
  const int be = tid >> 5;                  // element 0..15
  const int bl = lane;
  const int bi = bl & 15;                   // A row (output time within block)
  const int bK = be + (bl & 16);
  const int bu = bK & 15;                   // K within lag block
  const int bh = bK >> 4;                   // which lag block of the pair
  const int bdelta = causal ? (bi - bu) : (bu - bi);
  const int bwr = ((bl & 15) << 4) + ((bl >> 4) << 9)   // swizzled A write addr
                + ((be >> 3) << 8) + ((be & 7) << 1);

  // B constants: lane L holds col N = L&15 (= 4*deltaI + batch), K-half = L>>4
  const int col = lane & 15;
  const int hB  = lane >> 4;
  const int dI  = col >> 2;
  const int bb  = col & 3;
  const int ardA = ((lane & 15) << 4) + ((lane >> 4) << 9); // A read base

  // ------------------------------- main loop --------------------------------
  for (int R = 0; R < ROUNDS; ++R) {
    __syncthreads();                       // prev consumption done (and prologue)
    // cooperative build of 4 A-tiles: A[i, u+16h] = k[32*mb + 16h +/- (i-u)]
    #pragma unroll
    for (int s = 0; s < 4; ++s) {
      int kidx = 128 * R + 32 * s + 16 * bh + bdelta;
      _Float16 val = (kidx >= 0 && kidx < LLEN) ? kh[kidx] : (_Float16)0.0f;
      *(_Float16*)(aarena + (s << 10) + bwr) = val;
    }
    __syncthreads();                       // A-tiles ready

    #pragma unroll
    for (int s = 0; s < 4; ++s) {
      const int mb = 4 * R + s;            // lag-block-pair index
      // wave-uniform: does any of this wave's groups need this lag?
      bool any = causal ? (mb <= 2 * groups[3] + 1)
                        : (mb <= 255 - 2 * groups[0]);
      if (!any) continue;
      // hoisted shared A tile (loaded once, used by all active groups)
      union { v16h v; uint4 u[2]; } A;
      A.u[0] = *(const uint4*)(aarena + (s << 10) + ardA);
      A.u[1] = *(const uint4*)(aarena + (s << 10) + ardA + 256);
      #pragma unroll
      for (int j = 0; j < 4; ++j) {
        const int g = groups[j];
        bool active = causal ? (mb <= 2 * g + 1) : (mb <= 255 - 2 * g);
        if (active) {
          int J = causal ? (4 * g + dI - 2 * mb - hB)
                         : (4 * g + dI + 2 * mb + hB);
          union { v16h v; uint4 u[2]; } B;
          B.u[0] = (uint4){0, 0, 0, 0};
          B.u[1] = (uint4){0, 0, 0, 0};
          if (J >= 0 && J < NBLK) {
            int q  = (J << 2) | bb;
            int a0 = ((q & 15) << 4) + ((q >> 4) << 9);
            B.u[0] = *(const uint4*)(xarena + a0);
            B.u[1] = *(const uint4*)(xarena + a0 + 256);
          }
          acc[j] = __builtin_amdgcn_wmma_f32_16x16x32_f16(
              false, A.v, false, B.v, (short)0, acc[j], false, false);
        }
      }
    }
  }

  // ------------------------------- epilogue ---------------------------------
  // Restage each 16x16 D tile through wave-private LDS, then coalesced
  // residual loads + b128 stores: out = y + x * D[d].
  const float Dval = Dg[d];
  float* st = &stage[wv << 8];
  const int hl  = lane >> 4;
  const int ob  = lane >> 3;               // batch for the store phase
  const int opp = lane & 7;                // 8-float chunk within 64 times
  #pragma unroll
  for (int j = 0; j < 4; ++j) {
    const int g = groups[j];
    const int tbase = (g << 6) + (opp << 3);        // 64*g + 8*chunk
    const float* xr = x   + (size_t)ob * bstride + chan_off + tbase;
    float*       po = out + (size_t)ob * bstride + chan_off + tbase;
    __builtin_prefetch(xr, 0, 3);                   // global_prefetch_b8
    #pragma unroll
    for (int r = 0; r < 8; ++r) {
      // D element r at lane: M = r + 8*(lane>>4), N = lane&15
      st[((r + (hl << 3)) << 4) + col] = acc[j][r];
    }
    asm volatile("s_wait_dscnt 0x0" ::: "memory");  // cross-lane LDS exchange
    float4 r0 = ((const float4*)xr)[0];
    float4 r1 = ((const float4*)xr)[1];
    float res[8] = { r0.x, r0.y, r0.z, r0.w, r1.x, r1.y, r1.z, r1.w };
    float o[8];
    #pragma unroll
    for (int q = 0; q < 8; ++q) {
      int tl  = (opp << 3) + q;            // local time in 0..63
      int dI2 = tl >> 4;
      int i2  = tl & 15;
      o[q] = st[(i2 << 4) + (dI2 << 2) + ob] + res[q] * Dval;
    }
    float4 o0 = { o[0], o[1], o[2], o[3] };
    float4 o1 = { o[4], o[5], o[6], o[7] };
    ((float4*)po)[0] = o0;
    ((float4*)po)[1] = o1;
    asm volatile("s_wait_dscnt 0x0" ::: "memory");
  }
}

extern "C" void kernel_launch(void* const* d_in, const int* in_sizes, int n_in,
                              void* d_out, int out_size, void* d_ws, size_t ws_size,
                              hipStream_t stream) {
  (void)in_sizes; (void)n_in; (void)d_ws; (void)ws_size; (void)out_size;
  const float* x  = (const float*)d_in[0];
  const float* kk = (const float*)d_in[1];
  const float* Dg = (const float*)d_in[2];
  float* out = (float*)d_out;
  hipLaunchKernelGGL(hyena_tconv_wmma, dim3(2048), dim3(THREADS), 0, stream,
                     x, kk, Dg, out);
}
